// RVQVAEModel_62569083568269
// MI455X (gfx1250) — compile-verified, matrix operand order checked by reading, not run on hardware
//
#include <hip/hip_runtime.h>
#include <hip/hip_bf16.h>
#include <stdint.h>

// ---------------------------------------------------------------------------
// RVQ-VAE forward for MI455X (gfx1250, wave32, WMMA 16x16x32 f16->f32)
// GEMM mainloop: 128x128 workgroup tile, 8 waves (4x2), 32x64 per wave,
// double-buffered LDS fed by GLOBAL_LOAD_ASYNC_TO_LDS + s_wait_asynccnt.
// ---------------------------------------------------------------------------

typedef __attribute__((ext_vector_type(16))) _Float16 v16h;
typedef __attribute__((ext_vector_type(8)))  float    v8f;

#define B_   8
#define D_   512
#define T_   4096
#define KC_  1024
#define F_   2048
#define NQ_  8
#define KS_  7
#define BT_  (B_*T_)

#define RS_  40              // LDS row stride in halfs (80B, 16B aligned)
#define BUFH (128*RS_)       // halfs per LDS buffer

union Frag { v16h v; float4 q[2]; };

#if __has_builtin(__builtin_amdgcn_global_load_async_to_lds_b128)
#define ASYNC_LDS 1
typedef int v4i_ __attribute__((vector_size(16)));
typedef __attribute__((address_space(1))) v4i_ g_v4i;   // global (printed __device__)
typedef __attribute__((address_space(3))) v4i_ l_v4i;   // LDS (printed __shared__)
#else
#define ASYNC_LDS 0
#endif

static __device__ __forceinline__ void cp16(const _Float16* g, _Float16* l) {
#if ASYNC_LDS
  __builtin_amdgcn_global_load_async_to_lds_b128((g_v4i*)g, (l_v4i*)l, 0, 0);
#else
  *(float4*)l = *(const float4*)g;
#endif
}

template <int N>
static __device__ __forceinline__ void wait_async() {
#if ASYNC_LDS
# if __has_builtin(__builtin_amdgcn_s_wait_asynccnt)
  __builtin_amdgcn_s_wait_asynccnt(N);
# else
  asm volatile("s_wait_asynccnt %0" :: "i"(N) : "memory");
# endif
#endif
}

static __device__ __forceinline__ unsigned fkey(float f) {
  unsigned u = __float_as_uint(f);
  return (u & 0x80000000u) ? ~u : (u | 0x80000000u);   // monotonic float key
}

static __device__ __forceinline__ unsigned long long shfl_xor_u64(unsigned long long v, int m) {
  unsigned lo = (unsigned)v, hi = (unsigned)(v >> 32);
  lo = __shfl_xor(lo, m, 32);
  hi = __shfl_xor(hi, m, 32);
  return ((unsigned long long)hi << 32) | lo;
}

// issue cooperative copy of one K-tile (A: 128x32 halfs, W: 128x32 halfs)
// fixed trip count -> branch-free: exactly 4 async instructions per wave
static __device__ __forceinline__ void issue_tile(
    const _Float16* __restrict__ A, const _Float16* __restrict__ W, int Kdim,
    int m0, int n0, int k0, _Float16* sA, _Float16* sW)
{
  const int tid = threadIdx.x;
  #pragma unroll
  for (int it = 0; it < 2; ++it) {
    int c = tid + it*256;              // 0..511, no guard needed
    int r = c >> 2, ch = c & 3;
    cp16(A + (size_t)(m0 + r)*Kdim + k0 + ch*8, sA + r*RS_ + ch*8);
  }
  #pragma unroll
  for (int it = 0; it < 2; ++it) {
    int c = tid + it*256;
    int r = c >> 2, ch = c & 3;
    cp16(W + (size_t)(n0 + r)*Kdim + k0 + ch*8, sW + r*RS_ + ch*8);
  }
}

// A: [M,K] f16 row-major.  W: [N,K] f16 row-major (weights are [out,in]).
__device__ __forceinline__ void wmma_mainloop(
    const _Float16* __restrict__ A, const _Float16* __restrict__ W, int Kdim,
    int m0, int n0, _Float16* sA, _Float16* sW, v8f acc[2][4])
{
  const int lane = threadIdx.x & 31;
  const int wave = threadIdx.x >> 5;
  const int wm = wave >> 1, wn = wave & 1;
  const int h  = lane >> 4, lr = lane & 15;

  #pragma unroll
  for (int mi = 0; mi < 2; ++mi)
    #pragma unroll
    for (int ni = 0; ni < 4; ++ni)
      acc[mi][ni] = (v8f){0.f,0.f,0.f,0.f,0.f,0.f,0.f,0.f};

  int cur = 0;
  issue_tile(A, W, Kdim, m0, n0, 0, sA, sW);

  for (int k0 = 0; k0 < Kdim; k0 += 32) {
    bool more = (k0 + 32) < Kdim;
    if (more) {
      issue_tile(A, W, Kdim, m0, n0, k0 + 32,
                 sA + (cur ^ 1)*BUFH, sW + (cur ^ 1)*BUFH);
      wait_async<4>();           // tile k0 complete (in-order), k0+32 in flight
    } else {
      wait_async<0>();
    }
    __syncthreads();             // all waves' copies into buf[cur] visible

    const _Float16* cA = sA + cur*BUFH;
    const _Float16* cW = sW + cur*BUFH;
    Frag a[2], b[4];
    #pragma unroll
    for (int mi = 0; mi < 2; ++mi) {
      int row = wm*32 + mi*16 + lr;
      a[mi].q[0] = *(const float4*)(cA + row*RS_ + h*8);       // K = h*8..
      a[mi].q[1] = *(const float4*)(cA + row*RS_ + 16 + h*8);  // K = 16+h*8..
    }
    #pragma unroll
    for (int ni = 0; ni < 4; ++ni) {
      int row = wn*64 + ni*16 + lr;
      b[ni].q[0] = *(const float4*)(cW + row*RS_ + h*8);
      b[ni].q[1] = *(const float4*)(cW + row*RS_ + 16 + h*8);
    }
    #pragma unroll
    for (int mi = 0; mi < 2; ++mi)
      #pragma unroll
      for (int ni = 0; ni < 4; ++ni)
        acc[mi][ni] = __builtin_amdgcn_wmma_f32_16x16x32_f16(
            false, a[mi].v, false, b[ni].v, (short)0, acc[mi][ni], false, false);

    __syncthreads();             // done reading buf[cur] before reuse
    cur ^= 1;
  }
}

#define EPILOG_IDX \
  const int lane = threadIdx.x & 31, wave = threadIdx.x >> 5; \
  const int wm = wave >> 1, wn = wave & 1; \
  const int h = lane >> 4, lr = lane & 15; (void)wn; (void)lr;

// ---------------------------------------------------------------------------
// 1) causal depthwise conv + bias -> hconv f32 [B,D,T]
__global__ void k_dwconv(const float* __restrict__ x, const float* __restrict__ w,
                         const float* __restrict__ bias, float* __restrict__ out)
{
  size_t idx = (size_t)blockIdx.x*256 + threadIdx.x;   // over B*D*T
  int t = (int)(idx & 4095);
  int d = (int)((idx >> 12) & 511);
  const float* xp = x + (idx >> 12) * (size_t)T_;      // (b*D+d)*T
  float acc = bias[d];
  #pragma unroll
  for (int j = 0; j < KS_; ++j) {
    int tt = t - (KS_ - 1) + j;
    if (tt >= 0) acc += xp[tt] * w[d*KS_ + j];
  }
  out[idx] = acc;
}

// 2) channel LayerNorm -> hn f16 [BT, D]
__global__ __launch_bounds__(128) void k_ln(const float* __restrict__ hc,
    const float* __restrict__ w, const float* __restrict__ b, _Float16* __restrict__ hn)
{
  int m = blockIdx.x;                // b*T + t
  int bb = m >> 12, t = m & 4095;
  const float* src = hc + (size_t)bb*D_*T_ + t;
  float vals[4]; float s = 0.f, ss = 0.f;
  #pragma unroll
  for (int i = 0; i < 4; ++i) {
    int d = threadIdx.x + i*128;
    float v = src[(size_t)d*T_];
    vals[i] = v; s += v; ss += v*v;
  }
  __shared__ float rs[128], rq[128];
  rs[threadIdx.x] = s; rq[threadIdx.x] = ss; __syncthreads();
  for (int o = 64; o; o >>= 1) {
    if (threadIdx.x < o) { rs[threadIdx.x] += rs[threadIdx.x+o]; rq[threadIdx.x] += rq[threadIdx.x+o]; }
    __syncthreads();
  }
  float mean = rs[0] * (1.f/512.f);
  float var  = rq[0] * (1.f/512.f) - mean*mean;
  float rstd = rsqrtf(var + 1e-6f);
  #pragma unroll
  for (int i = 0; i < 4; ++i) {
    int d = threadIdx.x + i*128;
    float o = (vals[i] - mean) * rstd * w[d] + b[d];
    hn[(size_t)m*D_ + d] = (_Float16)o;
  }
}

// 3) generic f32 -> f16 convert
__global__ void k_cvt(const float* __restrict__ s, _Float16* __restrict__ d, int n) {
  int i = blockIdx.x*256 + threadIdx.x;
  if (i < n) d[i] = (_Float16)s[i];
}

// 4) codebook squared norms: rows = NQ*KC
__global__ void k_cbnorm(const float* __restrict__ cb, float* __restrict__ cn) {
  int r = blockIdx.x*256 + threadIdx.x;
  if (r >= NQ_*KC_) return;
  const float* c = cb + (size_t)r*D_;
  float s = 0.f;
  for (int d = 0; d < D_; ++d) s += c[d]*c[d];
  cn[r] = s;
}

// 5) pw1 GEMM + bias + gelu(tanh) -> f16 [BT, F]
__global__ __launch_bounds__(256) void k_gemm_pw1(
    const _Float16* __restrict__ A, const _Float16* __restrict__ W,
    const float* __restrict__ bias, _Float16* __restrict__ out)
{
  __shared__ __align__(16) _Float16 sA[2*BUFH];
  __shared__ __align__(16) _Float16 sW[2*BUFH];
  v8f acc[2][4];
  int m0 = blockIdx.x*128, n0 = blockIdx.y*128;
  wmma_mainloop(A, W, D_, m0, n0, sA, sW, acc);
  EPILOG_IDX
  #pragma unroll
  for (int mi = 0; mi < 2; ++mi)
    #pragma unroll
    for (int ni = 0; ni < 4; ++ni)
      #pragma unroll
      for (int v = 0; v < 8; ++v) {
        int grow = m0 + wm*32 + mi*16 + h*8 + v;
        int gcol = n0 + wn*64 + ni*16 + lr;
        float u = acc[mi][ni][v] + bias[gcol];
        float g = 0.5f*u*(1.f + tanhf(0.7978845608f*(u + 0.044715f*u*u*u)));
        out[(size_t)grow*F_ + gcol] = (_Float16)g;
      }
}

// 6) pw2 GEMM + bias + residual(x) -> res_f32/res_f16 (=z), qsum=0
__global__ __launch_bounds__(256) void k_gemm_pw2(
    const _Float16* __restrict__ A, const _Float16* __restrict__ W,
    const float* __restrict__ bias, const float* __restrict__ x,
    float* __restrict__ resf, _Float16* __restrict__ resh, float* __restrict__ qsum)
{
  __shared__ __align__(16) _Float16 sA[2*BUFH];
  __shared__ __align__(16) _Float16 sW[2*BUFH];
  v8f acc[2][4];
  int m0 = blockIdx.x*128, n0 = blockIdx.y*128;
  wmma_mainloop(A, W, F_, m0, n0, sA, sW, acc);
  EPILOG_IDX
  #pragma unroll
  for (int mi = 0; mi < 2; ++mi)
    #pragma unroll
    for (int ni = 0; ni < 4; ++ni)
      #pragma unroll
      for (int v = 0; v < 8; ++v) {
        int grow = m0 + wm*32 + mi*16 + h*8 + v;     // b*T + t
        int gcol = n0 + wn*64 + ni*16 + lr;          // channel d
        int bb = grow >> 12, t = grow & 4095;
        float z = acc[mi][ni][v] + bias[gcol] + x[((size_t)bb*D_ + gcol)*T_ + t];
        size_t o = (size_t)grow*D_ + gcol;
        resf[o] = z;
        resh[o] = (_Float16)z;
        qsum[o] = 0.f;
      }
}

// 7a) init per-row best (packed key|idx)
__global__ void k_init_best(unsigned long long* __restrict__ best) {
  int i = blockIdx.x*256 + threadIdx.x;
  if (i < BT_) best[i] = ~0ull;
}

// 7b) RVQ distance GEMM + fused argmin (lane-shuffle min + atomic_min_u64)
__global__ __launch_bounds__(256) void k_gemm_rvq(
    const _Float16* __restrict__ A, const _Float16* __restrict__ W,
    const float* __restrict__ cnorm, unsigned long long* __restrict__ best)
{
  __shared__ __align__(16) _Float16 sA[2*BUFH];
  __shared__ __align__(16) _Float16 sW[2*BUFH];
  v8f acc[2][4];
  int m0 = blockIdx.x*128, n0 = blockIdx.y*128;
  wmma_mainloop(A, W, D_, m0, n0, sA, sW, acc);
  EPILOG_IDX
  #pragma unroll
  for (int mi = 0; mi < 2; ++mi)
    #pragma unroll
    for (int v = 0; v < 8; ++v) {
      int grow = m0 + wm*32 + mi*16 + h*8 + v;
      unsigned long long p = ~0ull;
      #pragma unroll
      for (int ni = 0; ni < 4; ++ni) {
        int gcol = n0 + wn*64 + ni*16 + lr;
        float dist = cnorm[gcol] - 2.f*acc[mi][ni][v];
        unsigned long long q = ((unsigned long long)fkey(dist) << 32) | (unsigned)gcol;
        p = (q < p) ? q : p;
      }
      // min across the 16 columns held by lanes lr=0..15 (same row)
      #pragma unroll
      for (int o = 8; o; o >>= 1) {
        unsigned long long q = shfl_xor_u64(p, o);
        p = (q < p) ? q : p;
      }
      if (lr == 0) atomicMin(best + grow, p);
    }
}

// 7c) gather codeword, update residual (f32+f16), accumulate qsum, emit index
__global__ void k_rvq_update(const unsigned long long* __restrict__ best,
                             const float* __restrict__ cb,   // stage codebook f32 [KC,D]
                             float* __restrict__ resf, _Float16* __restrict__ resh,
                             float* __restrict__ qsum, int* __restrict__ idx_out)
{
  size_t i = (size_t)blockIdx.x*256 + threadIdx.x;  // over BT*D
  int m = (int)(i >> 9), d = (int)(i & 511);
  unsigned k = (unsigned)(best[m] & 0xffffffffull);
  float q = cb[(size_t)k*D_ + d];
  float r = resf[i] - q;
  resf[i] = r;
  resh[i] = (_Float16)r;
  qsum[i] += q;
  if (d == 0) idx_out[m] = (int)k;
}

// 8) decode GEMM + bias -> y [B, D, T]  (zq == qsum in forward)
__global__ __launch_bounds__(256) void k_gemm_dec(
    const _Float16* __restrict__ A, const _Float16* __restrict__ W,
    const float* __restrict__ bias, float* __restrict__ y)
{
  __shared__ __align__(16) _Float16 sA[2*BUFH];
  __shared__ __align__(16) _Float16 sW[2*BUFH];
  v8f acc[2][4];
  int m0 = blockIdx.x*128, n0 = blockIdx.y*128;
  wmma_mainloop(A, W, D_, m0, n0, sA, sW, acc);
  EPILOG_IDX
  #pragma unroll
  for (int mi = 0; mi < 2; ++mi)
    #pragma unroll
    for (int ni = 0; ni < 4; ++ni)
      #pragma unroll
      for (int v = 0; v < 8; ++v) {
        int grow = m0 + wm*32 + mi*16 + h*8 + v;     // b*T + t
        int gcol = n0 + wn*64 + ni*16 + lr;          // output channel o
        int bb = grow >> 12, t = grow & 4095;
        y[((size_t)bb*D_ + gcol)*T_ + t] = acc[mi][ni][v] + bias[gcol];
      }
}

// ---------------------------------------------------------------------------
extern "C" void kernel_launch(void* const* d_in, const int* in_sizes, int n_in,
                              void* d_out, int out_size, void* d_ws, size_t ws_size,
                              hipStream_t stream) {
  const float* x     = (const float*)d_in[0];
  const float* dw_w  = (const float*)d_in[1];
  const float* dw_b  = (const float*)d_in[2];
  const float* ln_w  = (const float*)d_in[3];
  const float* ln_b  = (const float*)d_in[4];
  const float* pw1_w = (const float*)d_in[5];
  const float* pw1_b = (const float*)d_in[6];
  const float* pw2_w = (const float*)d_in[7];
  const float* pw2_b = (const float*)d_in[8];
  const float* cbs   = (const float*)d_in[9];
  const float* dec_w = (const float*)d_in[10];
  const float* dec_b = (const float*)d_in[11];

  float* y_out   = (float*)d_out;
  int*   idx_out = (int*)d_out + (size_t)B_*D_*T_;

  // workspace carve-up (256B aligned)
  char* w = (char*)d_ws;
  auto carve = [&](size_t bytes) { void* p = (void*)w; w += (bytes + 255) & ~(size_t)255; return p; };
  float*    hconv  = (float*)   carve((size_t)BT_*D_*4);   // also res_f32 later
  _Float16* hn     = (_Float16*)carve((size_t)BT_*D_*2);   // also qsum_f16 later
  _Float16* out1   = (_Float16*)carve((size_t)BT_*F_*2);
  float*    qsum   = (float*)   carve((size_t)BT_*D_*4);
  _Float16* resh   = (_Float16*)carve((size_t)BT_*D_*2);
  _Float16* w1h    = (_Float16*)carve((size_t)F_*D_*2);
  _Float16* w2h    = (_Float16*)carve((size_t)D_*F_*2);
  _Float16* dech   = (_Float16*)carve((size_t)D_*D_*2);
  _Float16* cbh    = (_Float16*)carve((size_t)NQ_*KC_*D_*2);
  float*    cnorm  = (float*)   carve((size_t)NQ_*KC_*4);
  unsigned long long* best = (unsigned long long*)carve((size_t)BT_*8);
  float*    resf   = hconv;    // alias: hconv consumed before pw2 writes resf
  _Float16* qh     = hn;       // alias: hn consumed before qsum conversion

  // weight conversions (f32 -> f16)
  k_cvt<<<(F_*D_)/256, 256, 0, stream>>>(pw1_w, w1h, F_*D_);
  k_cvt<<<(D_*F_)/256, 256, 0, stream>>>(pw2_w, w2h, D_*F_);
  k_cvt<<<(D_*D_)/256, 256, 0, stream>>>(dec_w, dech, D_*D_);
  k_cvt<<<(NQ_*KC_*D_)/256, 256, 0, stream>>>(cbs, cbh, NQ_*KC_*D_);
  k_cbnorm<<<(NQ_*KC_)/256, 256, 0, stream>>>(cbs, cnorm);

  // ConvNeXt1d block
  k_dwconv<<<((size_t)B_*D_*T_)/256, 256, 0, stream>>>(x, dw_w, dw_b, hconv);
  k_ln<<<BT_, 128, 0, stream>>>(hconv, ln_w, ln_b, hn);
  k_gemm_pw1<<<dim3(BT_/128, F_/128), 256, 0, stream>>>(hn, w1h, pw1_b, out1);
  k_gemm_pw2<<<dim3(BT_/128, D_/128), 256, 0, stream>>>(out1, w2h, pw2_b, x, resf, resh, qsum);

  // residual VQ: 8 stages
  for (int s = 0; s < NQ_; ++s) {
    k_init_best<<<BT_/256, 256, 0, stream>>>(best);
    k_gemm_rvq<<<dim3(BT_/128, KC_/128), 256, 0, stream>>>(
        resh, cbh + (size_t)s*KC_*D_, cnorm + s*KC_, best);
    k_rvq_update<<<((size_t)BT_*D_)/256, 256, 0, stream>>>(
        best, cbs + (size_t)s*KC_*D_, resf, resh, qsum, idx_out + (size_t)s*BT_);
  }

  // decode: y = dec_w @ qsum + dec_b   (forward zq == qsum)
  k_cvt<<<((size_t)BT_*D_)/256, 256, 0, stream>>>(qsum, qh, BT_*D_);
  k_gemm_dec<<<dim3(BT_/128, D_/128), 256, 0, stream>>>(qh, dech, dec_b, y_out);
}
